// PTV3_Embedding_38371237822874
// MI455X (gfx1250) — matrix operand order repaired and checked by reading, not compiled, not run on hardware
//
#include <hip/hip_runtime.h>
#include <math.h>

#define NVOX_CIN 64
#define KOFF 27
#define CIN 64
#define COUT 128
#define NSTEPS (KOFF * 2)          // 54 K=32 steps
#define FRAG_ELEMS 512             // one B tile: 32 lanes x 16 bf16
#define STEP_ELEMS (8 * FRAG_ELEMS)// 8 col tiles per step = 4096 bf16 = 8KB

static constexpr float LN_EPS = 1e-5f;

typedef __attribute__((ext_vector_type(16))) __bf16 v16bf;
typedef __attribute__((ext_vector_type(8)))  float  v8f;

// ---------------------------------------------------------------------------
// Prologue: convert W [27][64][128] f32 -> pre-swizzled bf16 B fragments in ws.
// Layout: block b = (k*2+kk)*8 + j  (54 steps x 8 col tiles),
// thread t = lane*16 + e. Lane fragment is 16 contiguous bf16 (32B) so the
// main kernel reads it with a single (pair of) ds_load_b128.
// B element mapping (32x16, bf16): lanes 0-15 -> K={0..7,16..23}, lanes 16-31 -> K={8..15,24..31},
// N = lane & 15, element e -> K = (lane>>4)*8 + (e&7) + (e>>3)*16.
// ---------------------------------------------------------------------------
__global__ void swizzle_w_kernel(const float* __restrict__ W,
                                 __bf16* __restrict__ wsW) {
  const int b = blockIdx.x;       // 0..431
  const int t = threadIdx.x;      // 0..511
  const int j  = b & 7;
  const int kk = (b >> 3) & 1;
  const int k  = b >> 4;
  const int lane = t >> 4;
  const int e    = t & 15;
  const int Kfrag = ((lane >> 4) << 3) + (e & 7) + ((e >> 3) << 4); // 0..31
  const int cin   = kk * 32 + Kfrag;
  const int col   = j * 16 + (lane & 15);
  const float v = W[((size_t)k * CIN + cin) * COUT + col];
  wsW[(size_t)b * FRAG_ELEMS + t] = (__bf16)v;
}

// ---------------------------------------------------------------------------
// Main fused kernel: gather -> bf16 WMMA (f32 acc) -> LayerNorm -> exact GELU
// 256 threads = 8 waves, each wave owns 16 rows x 128 cols.
// ---------------------------------------------------------------------------
__global__ __launch_bounds__(256) void sparse_conv_ln_gelu_kernel(
    const float* __restrict__ feats, const int* __restrict__ nbr,
    const __bf16* __restrict__ wsW, const float* __restrict__ gamma,
    const float* __restrict__ beta, float* __restrict__ out, int n) {
  __shared__ __align__(32) __bf16 sB[STEP_ELEMS];   // 8 KB stage for one K-step

  const int lane  = threadIdx.x & 31;
  const int wave  = threadIdx.x >> 5;
  const int nIdx  = lane & 15;                 // N / M position within tile
  const int half  = lane >> 4;                 // lane group (0 or 1)
  const int chOff = half << 3;                 // A-fragment channel offset: 0 or 8
  const int rowBase = (blockIdx.x * 8 + wave) * 16;

  // Gather-index row for this lane (clamped so tail-block waves stay safe;
  // they still hit every barrier, only stores are predicated).
  int gRow = rowBase + nIdx;
  if (gRow >= n) gRow = n - 1;

  // gamma/beta for this lane's 8 columns (col = j*16 + nIdx)
  float gam[8], bet[8];
#pragma unroll
  for (int j = 0; j < 8; ++j) {
    gam[j] = gamma[j * 16 + nIdx];
    bet[j] = beta [j * 16 + nIdx];
  }

  v8f acc[8] = {};   // 16x16 f32 accumulators for col tiles j=0..7

  for (int k = 0; k < KOFF; ++k) {
    const int idx = nbr[(size_t)gRow * KOFF + k];
    const float* arow = feats + (size_t)idx * CIN;

#pragma unroll
    for (int kk = 0; kk < 2; ++kk) {
      const int step = k * 2 + kk;

      // ---- stage this step's B fragments (8 KB) into LDS, all 8 waves share
      __syncthreads();                               // previous step's reads done
      {
        const uint4* src = (const uint4*)(wsW + (size_t)step * STEP_ELEMS);
        uint4* dst = (uint4*)sB;
        dst[threadIdx.x]       = src[threadIdx.x];
        dst[threadIdx.x + 256] = src[threadIdx.x + 256];
      }
      if (step + 1 < NSTEPS)                         // warm next stage in L2
        __builtin_prefetch(wsW + (size_t)(step + 1) * STEP_ELEMS, 0, 1);
      __syncthreads();

      // ---- A fragment: gather 16 f32 channels for this lane, convert to bf16
      // ISA 16-bit A 16x32 layout: lane<16 holds K {0..7,16..23}, lane>=16
      // holds K {8..15,24..31}  (relative to kk*32).
      const float* ab = arow + kk * 32 + chOff;
      const float4 x = *(const float4*)(ab);
      const float4 y = *(const float4*)(ab + 4);
      const float4 z = *(const float4*)(ab + 16);
      const float4 w = *(const float4*)(ab + 20);
      v16bf A;
      A[0]  = (__bf16)x.x; A[1]  = (__bf16)x.y; A[2]  = (__bf16)x.z; A[3]  = (__bf16)x.w;
      A[4]  = (__bf16)y.x; A[5]  = (__bf16)y.y; A[6]  = (__bf16)y.z; A[7]  = (__bf16)y.w;
      A[8]  = (__bf16)z.x; A[9]  = (__bf16)z.y; A[10] = (__bf16)z.z; A[11] = (__bf16)z.w;
      A[12] = (__bf16)w.x; A[13] = (__bf16)w.y; A[14] = (__bf16)w.z; A[15] = (__bf16)w.w;

      // ---- 8 WMMAs: cols 0..127
#pragma unroll
      for (int j = 0; j < 8; ++j) {
        const v16bf Bj = *(const v16bf*)(sB + j * FRAG_ELEMS + lane * 16);
        acc[j] = __builtin_amdgcn_wmma_f32_16x16x32_bf16(
            /*neg_a=*/false, A, /*neg_b=*/false, Bj,
            /*c_mod=*/(short)0, acc[j], /*reuse_a=*/false, /*reuse_b=*/false);
      }
    }
  }

  // ---- LayerNorm + exact GELU epilogue, all in registers.
  // Accumulator layout: VGPR r, lanes 0-15 -> M=r, lanes 16-31 -> M=r+8.
  const float invC = 1.0f / (float)COUT;
#pragma unroll
  for (int r = 0; r < 8; ++r) {
    float s1 = 0.0f, s2 = 0.0f;
#pragma unroll
    for (int j = 0; j < 8; ++j) {
      const float v = acc[j][r];
      s1 += v; s2 += v * v;
    }
    // reduce across the 16 lanes of this half-group (masks 1,2,4,8 keep halves apart)
#pragma unroll
    for (int m = 1; m <= 8; m <<= 1) {
      s1 += __shfl_xor(s1, m, 32);
      s2 += __shfl_xor(s2, m, 32);
    }
    const float mu  = s1 * invC;
    const float var = s2 * invC - mu * mu;
    const float rs  = rsqrtf(var + LN_EPS);

    const int row = rowBase + r + (half << 3);
    if (row < n) {
      float* orow = out + (size_t)row * COUT;
#pragma unroll
      for (int j = 0; j < 8; ++j) {
        const float yv = (acc[j][r] - mu) * rs * gam[j] + bet[j];
        const float ge = 0.5f * yv * (1.0f + erff(yv * 0.70710678118654752f));
        orow[j * 16 + nIdx] = ge;
      }
    }
  }
}

// ---------------------------------------------------------------------------
extern "C" void kernel_launch(void* const* d_in, const int* in_sizes, int n_in,
                              void* d_out, int out_size, void* d_ws, size_t ws_size,
                              hipStream_t stream) {
  const float* feats = (const float*)d_in[0];  // [N, 64] f32
  const int*   nbr   = (const int*)  d_in[1];  // [N, 27] i32
  const float* W     = (const float*)d_in[2];  // [27, 64, 128] f32
  const float* gamma = (const float*)d_in[3];  // [128] f32
  const float* beta  = (const float*)d_in[4];  // [128] f32
  float* out = (float*)d_out;                  // [N, 128] f32
  const int n = in_sizes[0] / CIN;

  __bf16* wsW = (__bf16*)d_ws;                 // 54*8*512*2 B = 432 KB

  // 1) swizzle W into bf16 WMMA B-fragment layout
  swizzle_w_kernel<<<dim3(NSTEPS * 8), dim3(512), 0, stream>>>(W, wsW);

  // 2) fused gather-GEMM + LayerNorm + GELU
  const int blocks = (n + 127) / 128;          // 128 rows per block (8 waves)
  sparse_conv_ln_gelu_kernel<<<dim3(blocks), dim3(256), 0, stream>>>(
      feats, nbr, wsW, gamma, beta, out, n);
}
